// BaseConvLSTMRegressor_62491774156874
// MI455X (gfx1250) — compile-verified, matrix operand order checked by reading, not run on hardware
//
#include <hip/hip_runtime.h>
#include <hip/hip_bf16.h>

typedef __attribute__((ext_vector_type(16))) _Float16 v16h;
typedef __attribute__((ext_vector_type(8)))  float    v8f;

#define Bsz 16
#define Tt  24
#define Hh  96
#define Ww  96
#define Cc  8
#define Ff  16
#define KX   72    // 3*3*8  (input taps)
#define KTOT 216   // 72 + 3*3*16 (hidden taps)
#define KPAD 224   // 7 * 32
#define NOUT 64    // 4 gates * F
#define PIX  324   // 18*18 halo pixels

union AFrag { v16h v; uint4 q4[2]; unsigned u[8]; };
union CFrag { v8f  v; float f[8]; };
union CVT8  { _Float16 h[8]; uint4 q; };

// LDS element offset (f16 units) of even-K pair start, relative to the
// center-pixel base (pixel stride = 16 f16 = 32 B for BOTH regions).
// Region 0: x halo, ch 0-7 real, ch 8-15 always zero. Region 1: h halo.
// kt >= 216 (K padding) -> points at zero channels of region 0.
__host__ __device__ constexpr int deltaElem(int kt) {
    if (kt < KX) {
        int tap = kt / 8, c = kt % 8;
        return ((tap / 3 - 1) * 18 + (tap % 3 - 1)) * 16 + c;
    } else if (kt < KTOT) {
        int t2 = kt - KX;
        int tap = t2 / 16, c = t2 % 16;
        return PIX * 16 + ((tap / 3 - 1) * 18 + (tap % 3 - 1)) * 16 + c;
    }
    return 8; // zero pad: x-region center pixel, channels 8..15
}

__device__ __forceinline__ float hard_sigmoid(float x) {
    return fminf(fmaxf(0.2f * x + 0.5f, 0.0f), 1.0f);
}

// Pack w_x [3,3,8,64] and w_h [3,3,16,64] (fp32) into Wp2[n][kt] f16.
__global__ void prep_weights(const float* __restrict__ wx,
                             const float* __restrict__ wh,
                             _Float16* __restrict__ Wp2) {
    int idx = blockIdx.x * blockDim.x + threadIdx.x;
    if (idx >= NOUT * KPAD) return;
    int n = idx / KPAD, kt = idx % KPAD;
    float v = 0.f;
    if (kt < KX)        v = wx[(size_t)kt * NOUT + n];
    else if (kt < KTOT) v = wh[(size_t)(kt - KX) * NOUT + n];
    Wp2[(size_t)n * KPAD + kt] = (_Float16)v;
}

__global__ void init_state(_Float16* __restrict__ h0, float* __restrict__ c0, int n) {
    int i = blockIdx.x * blockDim.x + threadIdx.x;
    if (i < n) { h0[i] = (_Float16)0.f; c0[i] = 0.f; }
}

// One ConvLSTM timestep. Grid (6,6,16), block 256 (8 waves).
// Each WG: 16x16 pixel tile. Each wave: rows {2w, 2w+1} x 4 gate tiles.
__global__ __launch_bounds__(256)
void convlstm_step(const float* __restrict__ x,        // [B,T,H,W,C] fp32
                   int t,
                   const _Float16* __restrict__ h_prev, // [B,H,W,F] f16
                   const float* __restrict__ bias,      // [64]
                   const _Float16* __restrict__ Wp2,    // [64][224] f16
                   float* __restrict__ c_state,         // [B,H,W,F] f32 (in-place)
                   _Float16* __restrict__ h_next)       // [B,H,W,F] f16
{
    // Unified halo buffer: [region 0: x | region 1: h], 16 f16 per pixel each.
    __shared__ alignas(16) _Float16 sBuf[2 * PIX * 16];

    const int tX = blockIdx.x, tY = blockIdx.y, bb = blockIdx.z;
    const int tid = threadIdx.x;
    const int x0 = tX * 16, y0 = tY * 16;

    // ---- Stage x_t halo (one pixel per thread: 2 x b128 load, cvt, 2 x b128 store)
    for (int p = tid; p < PIX; p += 256) {
        int xx = p % 18, yy = p / 18;
        int gx = x0 + xx - 1, gy = y0 + yy - 1;
        bool inb = (gx >= 0) & (gx < Ww) & (gy >= 0) & (gy < Hh);
        float4 a = make_float4(0.f, 0.f, 0.f, 0.f);
        float4 b2 = make_float4(0.f, 0.f, 0.f, 0.f);
        if (inb) {
            const float* src = &x[((((size_t)bb * Tt + t) * Hh + gy) * Ww + gx) * Cc];
            a  = ((const float4*)src)[0];
            b2 = ((const float4*)src)[1];
            if (t + 1 < Tt)  // warm L2 for the next timestep's tile
                __builtin_prefetch((const void*)(src + (size_t)Hh * Ww * Cc), 0, 1);
        }
        CVT8 cv;
        cv.h[0] = (_Float16)a.x;  cv.h[1] = (_Float16)a.y;
        cv.h[2] = (_Float16)a.z;  cv.h[3] = (_Float16)a.w;
        cv.h[4] = (_Float16)b2.x; cv.h[5] = (_Float16)b2.y;
        cv.h[6] = (_Float16)b2.z; cv.h[7] = (_Float16)b2.w;
        *(uint4*)&sBuf[p * 16]     = cv.q;
        *(uint4*)&sBuf[p * 16 + 8] = make_uint4(0u, 0u, 0u, 0u); // zero channels
    }
    // ---- Stage h_prev halo (one pixel per thread: 2 x b128 copy)
    for (int p = tid; p < PIX; p += 256) {
        int xx = p % 18, yy = p / 18;
        int gx = x0 + xx - 1, gy = y0 + yy - 1;
        bool inb = (gx >= 0) & (gx < Ww) & (gy >= 0) & (gy < Hh);
        uint4 v0 = make_uint4(0u, 0u, 0u, 0u);
        uint4 v1 = make_uint4(0u, 0u, 0u, 0u);
        if (inb) {
            const uint4* src =
                (const uint4*)&h_prev[(((size_t)bb * Hh + gy) * Ww + gx) * Ff];
            v0 = src[0];
            v1 = src[1];
        }
        *(uint4*)&sBuf[PIX * 16 + p * 16]     = v0;
        *(uint4*)&sBuf[PIX * 16 + p * 16 + 8] = v1;
    }
    __syncthreads();

    const int lane = tid & 31;
    const int wave = tid >> 5;      // 0..7
    const int half = lane >> 4;     // K-half selector per WMMA layout
    const int ln   = lane & 15;     // A: pixel-in-row; B/D: channel
    const int row0 = wave * 2;

    CFrag acc[2][4];
    #pragma unroll
    for (int m = 0; m < 2; ++m)
        #pragma unroll
        for (int g = 0; g < 4; ++g)
            #pragma unroll
            for (int q = 0; q < 8; ++q) acc[m][g].f[q] = 0.f;

    // Hoisted bases: A center-pixel element offsets, B row pointers.
    const int pixE[2] = { ((row0 + 1) * 18 + ln + 1) * 16,
                          ((row0 + 2) * 18 + ln + 1) * 16 };
    const _Float16* wrow[4];
    #pragma unroll
    for (int g = 0; g < 4; ++g)
        wrow[g] = Wp2 + (size_t)(g * 16 + ln) * KPAD + half * 8;

    #pragma unroll
    for (int kk = 0; kk < 7; ++kk) {
        AFrag bf[4];
        #pragma unroll
        for (int g = 0; g < 4; ++g) {
            bf[g].q4[0] = *(const uint4*)(wrow[g] + kk * 32);
            bf[g].q4[1] = *(const uint4*)(wrow[g] + kk * 32 + 16);
        }
        AFrag af[2];
        #pragma unroll
        for (int m = 0; m < 2; ++m) {
            #pragma unroll
            for (int gi = 0; gi < 2; ++gi) {
                const int klo = kk * 32 + gi * 16;          // compile-time
                const int d = half ? deltaElem(klo + 8)     // const select
                                   : deltaElem(klo);
                af[m].q4[gi] = *(const uint4*)&sBuf[pixE[m] + d];
            }
        }
        #pragma unroll
        for (int m = 0; m < 2; ++m)
            #pragma unroll
            for (int g = 0; g < 4; ++g)
                acc[m][g].v = __builtin_amdgcn_wmma_f32_16x16x32_f16(
                    false, af[m].v, false, bf[g].v,
                    (short)0, acc[m][g].v, false, false);
    }

    // Gate math: D layout -> pixel M = v + 8*half (x within row), channel = ln
    const float bI = bias[0 * 16 + ln];
    const float bF = bias[1 * 16 + ln];
    const float bC = bias[2 * 16 + ln];
    const float bO = bias[3 * 16 + ln];

    #pragma unroll
    for (int m = 0; m < 2; ++m) {
        const int gy = y0 + row0 + m;
        #pragma unroll
        for (int v = 0; v < 8; ++v) {
            const int gx = x0 + v + half * 8;
            const size_t idx = (((size_t)bb * Hh + gy) * Ww + gx) * Ff + ln;
            float zi = acc[m][0].f[v] + bI;
            float zf = acc[m][1].f[v] + bF;
            float zc = acc[m][2].f[v] + bC;
            float zo = acc[m][3].f[v] + bO;
            float ig = hard_sigmoid(zi);
            float fg = hard_sigmoid(zf);
            float cn = fg * c_state[idx] + ig * fmaxf(zc, 0.f);
            float og = hard_sigmoid(zo);
            float hn = og * fmaxf(cn, 0.f);
            c_state[idx] = cn;
            h_next[idx]  = (_Float16)hn;
        }
    }
}

// Fused BatchNorm(inference) + Dense(1)
__global__ void bn_dense(const _Float16* __restrict__ hlast,
                         const float* __restrict__ gamma, const float* __restrict__ beta,
                         const float* __restrict__ mean,  const float* __restrict__ var,
                         const float* __restrict__ dw,    const float* __restrict__ db,
                         float* __restrict__ out, int npix)
{
    int p = blockIdx.x * blockDim.x + threadIdx.x;
    if (p >= npix) return;
    float s = 0.f;
    #pragma unroll
    for (int c = 0; c < Ff; ++c) {
        float inv = gamma[c] * rsqrtf(var[c] + 1e-3f);
        s += (float)hlast[(size_t)p * Ff + c] * (inv * dw[c])
           + (beta[c] - mean[c] * inv) * dw[c];
    }
    out[p] = s + db[0];
}

extern "C" void kernel_launch(void* const* d_in, const int* in_sizes, int n_in,
                              void* d_out, int out_size, void* d_ws, size_t ws_size,
                              hipStream_t stream) {
    const float* x     = (const float*)d_in[0];
    const float* wx    = (const float*)d_in[1];
    const float* wh    = (const float*)d_in[2];
    const float* bias  = (const float*)d_in[3];
    const float* gamma = (const float*)d_in[4];
    const float* beta  = (const float*)d_in[5];
    const float* mmean = (const float*)d_in[6];
    const float* mvar  = (const float*)d_in[7];
    const float* dw    = (const float*)d_in[8];
    const float* db    = (const float*)d_in[9];
    float* out = (float*)d_out;

    const size_t NST = (size_t)Bsz * Hh * Ww * Ff;  // 2,359,296 state elements
    char* ws = (char*)d_ws;
    _Float16* hA  = (_Float16*)ws;
    _Float16* hB  = hA + NST;
    float*    cs  = (float*)(ws + 2 * NST * sizeof(_Float16));
    _Float16* Wp2 = (_Float16*)(ws + 2 * NST * sizeof(_Float16) + NST * sizeof(float));

    prep_weights<<<(NOUT * KPAD + 255) / 256, 256, 0, stream>>>(wx, wh, Wp2);
    init_state<<<((int)NST + 255) / 256, 256, 0, stream>>>(hA, cs, (int)NST);

    dim3 grid(Ww / 16, Hh / 16, Bsz);
    _Float16* hp = hA;
    _Float16* hn = hB;
    for (int t = 0; t < Tt; ++t) {
        convlstm_step<<<grid, 256, 0, stream>>>(x, t, hp, bias, Wp2, cs, hn);
        _Float16* tmp = hp; hp = hn; hn = tmp;
    }

    const int npix = Bsz * Hh * Ww;
    bn_dense<<<(npix + 255) / 256, 256, 0, stream>>>(hp, gamma, beta, mmean, mvar,
                                                     dw, db, out, npix);
}
